// GraphIsomorphism_Embs_89094801588703
// MI455X (gfx1250) — matrix operand order repaired and checked by reading, not compiled, not run on hardware
//
#include <hip/hip_runtime.h>
#include <hip/hip_bf16.h>

// ---------------------------------------------------------------------------
// CDNA5 / gfx1250 GIN pipeline.
// GEMMs run on v_wmma_f32_16x16x32_bf16 (wave32, 16x16 f32 accumulators),
// double-buffered via a fully-unrolled k-loop (stage swap = register renaming,
// no v_mov rotation), wave tile 32x64 for B-fragment reuse.
// ---------------------------------------------------------------------------

#define HDIM 512
#define NSRC 80000
#define NTGT 80000
#define NEDGE 400000
#define NLBL 200000

typedef __attribute__((ext_vector_type(16))) __bf16 v16bf;
typedef __attribute__((ext_vector_type(8)))  __bf16 v8bf;
typedef __attribute__((ext_vector_type(8)))  float  v8f;

__device__ __forceinline__ unsigned short f2bf(float f) {
    unsigned u = __float_as_uint(f);
    unsigned r = u + 0x7FFFu + ((u >> 16) & 1u);   // round-to-nearest-even
    return (unsigned short)(r >> 16);
}

__device__ __forceinline__ float lrelu(float x) {
    return x > 0.0f ? x : 0.01f * x;
}

// ---------------------------------------------------------------------------
// Weight conversion: 8 matrices W[k][n] f32 -> Wt[n][k] bf16 (B-fragment wants
// a column of W as 16 contiguous bf16 per 32-wide K step).
// ---------------------------------------------------------------------------
struct WPtrs { const float* p[8]; };

__global__ __launch_bounds__(256) void convw_kernel(WPtrs wp, unsigned short* __restrict__ dst) {
    unsigned tid = blockIdx.x * 256u + threadIdx.x;       // 8 * 512 * 512 total
    int m = tid >> 18;                                    // matrix slot
    unsigned r = tid & 262143u;
    int k = r >> 9;
    int n = r & 511;
    dst[((size_t)m << 18) + ((size_t)n << 9) + k] = f2bf(wp.p[m][((size_t)k << 9) + n]);
}

// ---------------------------------------------------------------------------
// Zero two f32 buffers (avoid hipMemsetAsync inside graph capture).
// ---------------------------------------------------------------------------
__global__ __launch_bounds__(256) void zero2_kernel(float* __restrict__ a, float* __restrict__ b, long n4) {
    long i = (long)blockIdx.x * blockDim.x + threadIdx.x;
    if (i >= n4) return;
    float4 z = make_float4(0.f, 0.f, 0.f, 0.f);
    ((float4*)a)[i] = z;
    ((float4*)b)[i] = z;
}

// ---------------------------------------------------------------------------
// Scatter-add aggregation: for edge e, agg_s[src] += xt[dst], agg_t[dst] += xs[src].
// One thread per (edge, 4-channel chunk): float4 gather + 8 global f32 atomics.
// ---------------------------------------------------------------------------
__global__ __launch_bounds__(256) void scatter_kernel(
    const int* __restrict__ src, const int* __restrict__ dst,
    const float* __restrict__ xs, const float* __restrict__ xt,
    float* __restrict__ agg_s, float* __restrict__ agg_t, long total) {
    long tid = (long)blockIdx.x * blockDim.x + threadIdx.x;
    if (tid >= total) return;
    long e = tid >> 7;                 // 128 chunks per edge
    int  c = (int)(tid & 127) << 2;    // channel base
    int  s = src[e], d = dst[e];
    float4 vt = *(const float4*)(xt + (size_t)d * HDIM + c);
    float4 vs = *(const float4*)(xs + (size_t)s * HDIM + c);
    float* ps = agg_s + (size_t)s * HDIM + c;
    float* pt = agg_t + (size_t)d * HDIM + c;
    atomicAdd(ps + 0, vt.x); atomicAdd(ps + 1, vt.y);
    atomicAdd(ps + 2, vt.z); atomicAdd(ps + 3, vt.w);
    atomicAdd(pt + 0, vs.x); atomicAdd(pt + 1, vs.y);
    atomicAdd(pt + 2, vs.z); atomicAdd(pt + 3, vs.w);
}

// ---------------------------------------------------------------------------
// Pre: y = (1+eps)*x + agg  -> bf16 row-major
// ---------------------------------------------------------------------------
__global__ __launch_bounds__(256) void pre_kernel(
    const float* __restrict__ x, const float* __restrict__ agg,
    const float* __restrict__ epsp, unsigned short* __restrict__ ybf, long n4) {
    long i = (long)blockIdx.x * blockDim.x + threadIdx.x;
    if (i >= n4) return;
    float e = 1.0f + *epsp;
    float4 xv = ((const float4*)x)[i];
    float4 av = ((const float4*)agg)[i];
    ushort4 o;
    o.x = f2bf(e * xv.x + av.x);
    o.y = f2bf(e * xv.y + av.y);
    o.z = f2bf(e * xv.z + av.z);
    o.w = f2bf(e * xv.w + av.w);
    ((ushort4*)ybf)[i] = o;
}

// ---------------------------------------------------------------------------
// WMMA GEMM: out[M,512] f32 = Abf[M,512](bf16 rows) x W (via Wt[n][k] bf16) + bias
// Block: 256 threads = 8 waves; wave computes 32 rows x 64 cols (8 accum tiles,
// two A fragments sharing four B fragments). Grid: (512/64, ceil(M/256)).
// K loop: 16 steps of 32, FULLY UNROLLED with 2-stage ping-pong fragments:
// the stage index (i & 1) is a compile-time constant, so the double-buffer
// rotation is pure register renaming (no v_mov chains), and loads for stage
// i+1 issue before the 8 WMMAs of stage i.
//
// Fragment layouts (wave32, v_wmma_f32_16x16x32_bf16; A/B use this code's contiguous k order):
//   A 16x32: lane l -> row m=l%16; K = kb+0..7 (v0..v3) and kb+8..15 (v4..v7),
//            kb = 16 if l>=16 else 0   -> two contiguous 16B loads.
//   B 32x16: lane l -> col n=l%16; K = 16*(l>=16) + 0..15, V_i = pair {2i,2i+1}
//            -> 16 contiguous bf16 of Wt[n] -> one 32B load.
//   C/D 16x16 f32: VGPR j, lanes 0-15 -> (row j, col lane); lanes 16-31 -> (row j+8).
// ---------------------------------------------------------------------------
__global__ __launch_bounds__(256) void gemm_bf16_kernel(
    const unsigned short* __restrict__ Abf,   // [M,512] bf16
    const unsigned short* __restrict__ Wt,    // [512,512] bf16, Wt[n*512+k]
    const float* __restrict__ bias,           // [512]
    float* __restrict__ out,                  // [M,512] f32
    int M) {
    const int lane  = threadIdx.x & 31;
    const int wave  = threadIdx.x >> 5;
    const int mtile = blockIdx.y * 256 + wave * 32;   // 32 rows per wave
    const int nbase = blockIdx.x * 64;                // 64 cols, 4 tiles
    if (mtile >= M) return;   // M % 32 == 0 -> waves are fully in or fully out

    const int lhalf = lane & 15;
    const int hi    = lane >> 4;          // 0 | 1
    const int akoff = hi * 16;

    const unsigned short* arow0 = Abf + (size_t)(mtile + lhalf) * HDIM;
    const unsigned short* arow1 = arow0 + (size_t)16 * HDIM;
    const unsigned short* wrow[4];
#pragma unroll
    for (int t = 0; t < 4; ++t)
        wrow[t] = Wt + (size_t)(nbase + t * 16 + lhalf) * HDIM + hi * 16;

    union AF { v16bf v; v8bf h[2]; };
    v8f acc[8] = {};
    AF a0[2], a1[2];
    v16bf bb[2][4];

    auto loadStage = [&](int st, int ks) {
        a0[st].h[0] = *(const v8bf*)(arow0 + ks + akoff);
        a0[st].h[1] = *(const v8bf*)(arow0 + ks + akoff + 8);
        a1[st].h[0] = *(const v8bf*)(arow1 + ks + akoff);
        a1[st].h[1] = *(const v8bf*)(arow1 + ks + akoff + 8);
#pragma unroll
        for (int t = 0; t < 4; ++t) bb[st][t] = *(const v16bf*)(wrow[t] + ks);
    };

    loadStage(0, 0);
#pragma unroll
    for (int i = 0; i < HDIM / 32; ++i) {       // 16 steps, fully unrolled
        const int cur = i & 1;
        if (i + 1 < HDIM / 32) loadStage(cur ^ 1, (i + 1) * 32);
#pragma unroll
        for (int t = 0; t < 4; ++t) {
            acc[t] = __builtin_amdgcn_wmma_f32_16x16x32_bf16(
                false, a0[cur].v, false, bb[cur][t], (short)0, acc[t], false, false);
        }
#pragma unroll
        for (int t = 0; t < 4; ++t) {
            acc[4 + t] = __builtin_amdgcn_wmma_f32_16x16x32_bf16(
                false, a1[cur].v, false, bb[cur][t], (short)0, acc[4 + t], false, false);
        }
    }

#pragma unroll
    for (int t = 0; t < 4; ++t) {
        const int col = nbase + t * 16 + lhalf;
        const float b = bias[col];
#pragma unroll
        for (int half = 0; half < 2; ++half) {
            const int rbase = mtile + half * 16 + 8 * hi;
#pragma unroll
            for (int j = 0; j < 8; ++j) {
                out[(size_t)(rbase + j) * HDIM + col] = acc[half * 4 + t][j] + b;
            }
        }
    }
}

// ---------------------------------------------------------------------------
// LayerNorm + LeakyReLU, one 128-thread block per row, output bf16.
// ---------------------------------------------------------------------------
__global__ __launch_bounds__(128) void ln_leaky_kernel(
    const float* __restrict__ z, const float* __restrict__ gam,
    const float* __restrict__ bet, unsigned short* __restrict__ obf) {
    const int row = blockIdx.x, t = threadIdx.x;
    float4 v = ((const float4*)(z + (size_t)row * HDIM))[t];
    float s = v.x + v.y + v.z + v.w;
    float q = v.x * v.x + v.y * v.y + v.z * v.z + v.w * v.w;
#pragma unroll
    for (int off = 16; off; off >>= 1) {
        s += __shfl_xor(s, off, 32);
        q += __shfl_xor(q, off, 32);
    }
    __shared__ float rs[4], rq[4];
    const int wv = t >> 5, ln = t & 31;
    if (ln == 0) { rs[wv] = s; rq[wv] = q; }
    __syncthreads();
    s = rs[0] + rs[1] + rs[2] + rs[3];
    q = rq[0] + rq[1] + rq[2] + rq[3];
    const float mu  = s * (1.0f / HDIM);
    const float var = q * (1.0f / HDIM) - mu * mu;
    const float r   = rsqrtf(var + 1e-5f);
    float4 g4 = ((const float4*)gam)[t];
    float4 b4 = ((const float4*)bet)[t];
    ushort4 o;
    o.x = f2bf(lrelu((v.x - mu) * r * g4.x + b4.x));
    o.y = f2bf(lrelu((v.y - mu) * r * g4.y + b4.y));
    o.z = f2bf(lrelu((v.z - mu) * r * g4.z + b4.z));
    o.w = f2bf(lrelu((v.w - mu) * r * g4.w + b4.w));
    ((ushort4*)(obf + (size_t)row * HDIM))[t] = o;
}

// ---------------------------------------------------------------------------
// Post: in-place l2-normalize a row, optional leaky_relu.
// ---------------------------------------------------------------------------
__global__ __launch_bounds__(128) void post_kernel(float* __restrict__ h, int leaky) {
    const int row = blockIdx.x, t = threadIdx.x;
    float4* p = (float4*)(h + (size_t)row * HDIM);
    float4 v = p[t];
    float q = v.x * v.x + v.y * v.y + v.z * v.z + v.w * v.w;
#pragma unroll
    for (int off = 16; off; off >>= 1) q += __shfl_xor(q, off, 32);
    __shared__ float rq[4];
    const int wv = t >> 5, ln = t & 31;
    if (ln == 0) rq[wv] = q;
    __syncthreads();
    q = rq[0] + rq[1] + rq[2] + rq[3];
    const float d = 1.0f / fmaxf(sqrtf(q), 1e-12f);
    v.x *= d; v.y *= d; v.z *= d; v.w *= d;
    if (leaky) { v.x = lrelu(v.x); v.y = lrelu(v.y); v.z = lrelu(v.z); v.w = lrelu(v.w); }
    p[t] = v;
}

// ---------------------------------------------------------------------------
// Final classifier: one wave (32 lanes) per labeled pair.
// ---------------------------------------------------------------------------
__global__ __launch_bounds__(256) void dot_kernel(
    const float* __restrict__ hs, const float* __restrict__ ht,
    const int* __restrict__ eli, float* __restrict__ out, int L) {
    const long tid = (long)blockIdx.x * blockDim.x + threadIdx.x;
    const long w = tid >> 5;
    const int lane = (int)(tid & 31);
    if (w >= L) return;
    const int i0 = eli[w], i1 = eli[L + w];
    const float4* a = (const float4*)(hs + (size_t)i0 * HDIM);
    const float4* b = (const float4*)(ht + (size_t)i1 * HDIM);
    float s = 0.f;
#pragma unroll
    for (int r = 0; r < 4; ++r) {
        float4 x = a[lane + 32 * r];
        float4 y = b[lane + 32 * r];
        s += x.x * y.x + x.y * y.y + x.z * y.z + x.w * y.w;
    }
#pragma unroll
    for (int off = 16; off; off >>= 1) s += __shfl_xor(s, off, 32);
    if (lane == 0) out[w] = s;
}

// ---------------------------------------------------------------------------
// Host orchestration
// ---------------------------------------------------------------------------
extern "C" void kernel_launch(void* const* d_in, const int* in_sizes, int n_in,
                              void* d_out, int out_size, void* d_ws, size_t ws_size,
                              hipStream_t stream) {
    const int* EI  = (const int*)d_in[0];        // [2,E]: src = EI, dst = EI+E
    const int* ELI = (const int*)d_in[1];        // [2,L]
    const float* emb_s = (const float*)d_in[2];
    const float* emb_t = (const float*)d_in[3];
    // groups: 0=1s, 1=1t, 2=2s, 3=2t ; fields: 0 eps, 1 A, 2 a, 3 g, 4 be, 5 B, 6 c
    auto G = [&](int g, int j) -> const float* { return (const float*)d_in[4 + g * 7 + j]; };

    const size_t NH  = (size_t)NSRC * HDIM;      // 40,960,000 elements
    char* ws = (char*)d_ws;
    float* F0    = (float*)ws; ws += NH * 4;     // source activations / output
    float* F1    = (float*)ws; ws += NH * 4;     // target activations / output
    float* AGG_S = (float*)ws; ws += NH * 4;     // agg (then reused as Z for s-MLP)
    float* AGG_T = (float*)ws; ws += NH * 4;     // agg (then reused as Z for t-MLP)
    unsigned short* Ybf = (unsigned short*)ws; ws += NH * 2;  // bf16 staging (pre & LN out)
    unsigned short* Wbf = (unsigned short*)ws;   // 8 x 512 x 512 bf16 transposed weights

    // One-time (per launch) weight conversion; slot g*2 = A, g*2+1 = B.
    WPtrs wp;
    for (int g = 0; g < 4; ++g) { wp.p[g * 2] = G(g, 1); wp.p[g * 2 + 1] = G(g, 5); }
    convw_kernel<<<8192, 256, 0, stream>>>(wp, Wbf);

    const long n4      = (long)(NH / 4);
    const int  zblocks = (int)((n4 + 255) / 256);
    const long sc_tot  = (long)NEDGE * (HDIM / 4);
    const int  scblocks = (int)((sc_tot + 255) / 256);
    const dim3 ggrid(HDIM / 64, (NSRC + 255) / 256);   // 8 x 313, wave-granular tail

    auto run_mlp = [&](const float* x, float* agg_then_z, int g, float* out, int leaky) {
        pre_kernel<<<zblocks, 256, 0, stream>>>(x, agg_then_z, G(g, 0), Ybf, n4);
        gemm_bf16_kernel<<<ggrid, 256, 0, stream>>>(
            Ybf, Wbf + (size_t)(g * 2) * HDIM * HDIM, G(g, 2), agg_then_z, NSRC);
        ln_leaky_kernel<<<NSRC, 128, 0, stream>>>(agg_then_z, G(g, 3), G(g, 4), Ybf);
        gemm_bf16_kernel<<<ggrid, 256, 0, stream>>>(
            Ybf, Wbf + (size_t)(g * 2 + 1) * HDIM * HDIM, G(g, 6), out, NSRC);
        post_kernel<<<NSRC, 128, 0, stream>>>(out, leaky);
    };

    auto run_layer = [&](const float* xs, const float* xt, int gs, int gt, int leaky) {
        zero2_kernel<<<zblocks, 256, 0, stream>>>(AGG_S, AGG_T, n4);
        scatter_kernel<<<scblocks, 256, 0, stream>>>(EI, EI + NEDGE, xs, xt, AGG_S, AGG_T, sc_tot);
        run_mlp(xs, AGG_S, gs, F0, leaky);
        run_mlp(xt, AGG_T, gt, F1, leaky);
    };

    run_layer(emb_s, emb_t, 0, 1, 1);   // layer 1: leaky(l2norm(.))
    run_layer(F0, F1, 2, 3, 0);         // layer 2: l2norm(.)

    dot_kernel<<<(NLBL * 32 + 255) / 256, 256, 0, stream>>>(F0, F1, ELI, (float*)d_out, NLBL);
}
